// Simple_TensorProduct_oTchannel_89979564851601
// MI455X (gfx1250) — matrix operand order repaired and checked by reading, not compiled, not run on hardware
//
#include <hip/hip_runtime.h>
#include <stdint.h>

typedef __attribute__((ext_vector_type(16))) _Float16 v16h;
typedef __attribute__((ext_vector_type(8)))  _Float16 v8h;
typedef __attribute__((ext_vector_type(4)))  _Float16 v4h;
typedef __attribute__((ext_vector_type(8)))  float    v8f;
typedef __attribute__((ext_vector_type(4)))  float    v4f;

#define NT 32            // nodes per workgroup tile
#define AS_STRIDE 136    // halves (272B rows: 16B aligned, bank-skewed)
#define BS_STRIDE 136
#define CSTRIDE   180    // floats per node for c-coeffs (179 used)

// ---- path metadata (INSTR order from the reference) ----
static constexpr int P_L1[15] = {0,0,0,1,1,1,1,1,1,2,2,2,2,2,2};
static constexpr int P_L2[15] = {0,1,2,0,1,1,1,2,2,0,1,1,2,2,2};
static constexpr int P_LO[15] = {0,1,2,1,0,1,2,1,2,2,1,2,0,1,2};
static constexpr int LBASE[3] = {0,1,4};
static constexpr int W3JOFF[15] = {0,1,10,35,44,53,80,125,170,245,270,315,390,415,490}; // i*j*k sizes cumsum
static constexpr int COFF[15]   = {0,1,4,9,18,21,30,45,54,69,94,109,134,139,154};       // k*i sizes cumsum (tot 179)
// groups by output block: io=0 (k=1), io=1 (k=3), io=2 (k=5)
static constexpr int GPATHS[3][6] = {{0,4,12,0,0,0},{1,3,5,7,10,13},{2,6,8,9,11,14}};

// =====================================================================
// Kernel 1: compute real-basis Wigner 3j tensors (e3nn convention) in
// fp64 on device, Frobenius-normalized, with path alpha folded in.
// =====================================================================
__device__ double dfact(int n) { double r = 1.0; for (int i = 2; i <= n; ++i) r *= (double)i; return r; }

__device__ double su2_cg(int j1, int m1, int j2, int m2, int j3, int m3) {
  if (m3 != m1 + m2) return 0.0;
  int vmin = -j1 + j2 + m3;
  if (-j1 + m1 > vmin) vmin = -j1 + m1;
  if (vmin < 0) vmin = 0;
  int vmax = j2 + j3 + m1;
  if (j3 - j1 + j2 < vmax) vmax = j3 - j1 + j2;
  if (j3 + m3 < vmax) vmax = j3 + m3;
  if (vmax < vmin) return 0.0;
  double C = sqrt((2.0*j3+1.0) * dfact(j3+j1-j2) * dfact(j3-j1+j2) * dfact(j1+j2-j3) / dfact(j1+j2+j3+1)
                  * dfact(j3+m3) * dfact(j3-m3)
                  / (dfact(j1-m1) * dfact(j1+m1) * dfact(j2-m2) * dfact(j2+m2)));
  double S = 0.0;
  for (int v = vmin; v <= vmax; ++v) {
    double sgn = ((v + j2 + m2) & 1) ? -1.0 : 1.0;
    S += sgn / dfact(v) * dfact(j2+j3+m1-v) * dfact(j1-m1+v)
         / (dfact(j3-j1+j2-v) * dfact(j3+m3-v) * dfact(v+j1-j2-m3));
  }
  return C * S;
}

__device__ void build_q(int l, double qr[5][5], double qi[5][5]) {
  for (int a = 0; a < 5; ++a)
    for (int b = 0; b < 5; ++b) { qr[a][b] = 0.0; qi[a][b] = 0.0; }
  const double s = 0.7071067811865475244;
  for (int m = -l; m < 0; ++m) {
    qr[l+m][l-m] = s;      // 1/sqrt2 at col l+|m|
    qi[l+m][l+m] = -s;     // -i/sqrt2 at col l-|m|
  }
  qr[l][l] = 1.0;
  for (int m = 1; m <= l; ++m) {
    double sg = (m & 1) ? -1.0 : 1.0;
    qr[l+m][l+m] = sg * s;
    qi[l+m][l-m] = sg * s;
  }
  // multiply by (-i)^l
  if (l == 1) {
    for (int a = 0; a < 5; ++a) for (int b = 0; b < 5; ++b) {
      double r = qr[a][b], im = qi[a][b];
      qr[a][b] = im; qi[a][b] = -r;      // (-i)*(r+i m) = m - i r
    }
  } else if (l == 2) {
    for (int a = 0; a < 5; ++a) for (int b = 0; b < 5; ++b) { qr[a][b] = -qr[a][b]; qi[a][b] = -qi[a][b]; }
  }
}

__global__ void w3j_prep(float* __restrict__ w3j_s) {
  int p = threadIdx.x;
  if (p >= 15) return;
  const int l1 = P_L1[p], l2 = P_L2[p], l3 = P_LO[p];
  const int d1 = 2*l1+1, d2 = 2*l2+1, d3 = 2*l3+1;
  double q1r[5][5], q1i[5][5], q2r[5][5], q2i[5][5], q3r[5][5], q3i[5][5];
  build_q(l1, q1r, q1i);
  build_q(l2, q2r, q2i);
  build_q(l3, q3r, q3i);
  double res[125];
  double nrm2 = 0.0;
  for (int a = 0; a < d1; ++a)
    for (int b = 0; b < d2; ++b)
      for (int c = 0; c < d3; ++c) {
        double acc = 0.0;
        for (int i = 0; i < d1; ++i)
          for (int k = 0; k < d2; ++k)
            for (int n = 0; n < d3; ++n) {
              double cg = su2_cg(l1, i-l1, l2, k-l2, l3, n-l3);
              if (cg == 0.0) continue;
              // Re( Q1[i][a] * Q2[k][b] * conj(Q3[n][c]) )
              double ar = q1r[i][a]*q2r[k][b] - q1i[i][a]*q2i[k][b];
              double ai = q1r[i][a]*q2i[k][b] + q1i[i][a]*q2r[k][b];
              acc += (ar * q3r[n][c] + ai * q3i[n][c]) * cg;  // t3i = -q3i
            }
        res[(a*d2 + b)*d3 + c] = acc;
        nrm2 += acc * acc;
      }
  const double cnt[3] = {3.0, 6.0, 6.0};  // paths feeding each output block
  double alpha = sqrt((2.0*l3 + 1.0) / (128.0 * cnt[l3]));
  double inv = alpha / sqrt(nrm2);
  const int tot = d1*d2*d3;
  for (int t = 0; t < tot; ++t) w3j_s[W3JOFF[p] + t] = (float)(res[t] * inv);
}

// =====================================================================
// Kernel 2: weights fp32 [p][u][w] -> f16 transposed [p][w][u]
// =====================================================================
__global__ void prep_weights(const float* __restrict__ w, _Float16* __restrict__ wt16) {
  int idx = blockIdx.x * blockDim.x + threadIdx.x;
  if (idx >= 15 * 16384) return;
  int p = idx >> 14;
  int rem = idx & 16383;
  int wc = rem >> 7;   // output channel
  int u  = rem & 127;  // input channel
  wt16[idx] = (_Float16)w[p * 16384 + u * 128 + wc];
}

// =====================================================================
// Kernel 3: fused tensor-product GEMM via v_wmma_f32_16x16x32_f16
// =====================================================================
template <int KG, int IOB, int NP, int G>
__device__ __forceinline__ void do_group(
    const float* __restrict__ x1, const _Float16* __restrict__ wt16,
    float* __restrict__ out, int Nn, int z0,
    _Float16* As, _Float16* Bs, const float* Cs)
{
  constexpr int MT = (NT * KG) / 16;   // M tiles of 16 rows
  const int tid  = threadIdx.x;
  const int lane = tid & 31;
  const int wv   = tid >> 5;           // wave id 0..7 -> 16-wide output column strip
  const int half = lane >> 4;
  const int l16  = lane & 15;

  v8f zero = {};
  v4f zero4 = {};
  v8f acc[MT];
#pragma unroll
  for (int m = 0; m < MT; ++m) acc[m] = zero;

#pragma unroll
  for (int pi = 0; pi < NP; ++pi) {
    const int p  = GPATHS[G][pi];
    const int l1 = P_L1[p];
    const int id = 2 * l1 + 1;
    const int s1 = LBASE[l1];
    const int co = COFF[p];

    // ---- stage B: async DMA W_p^T (f16) -> LDS, tracked by ASYNCcnt ----
    {
      const _Float16* wp = wt16 + p * 16384;
#pragma unroll
      for (int it = 0; it < 8; ++it) {
        int c8 = tid + it * 256;          // chunk of 8 halves, 0..2047
        int wrow = c8 >> 4;               // 0..127
        int u8 = (c8 & 15) * 8;
        uint32_t lds  = (uint32_t)(uintptr_t)&Bs[wrow * BS_STRIDE + u8];
        uint32_t goff = (uint32_t)((wrow * 128 + u8) * 2);   // byte offset in weight tile
        asm volatile("global_load_async_to_lds_b128 %0, %1, %2"
                     :: "v"(lds), "v"(goff), "s"(wp) : "memory");
      }
      // prefetch next path's weight tile into cache while this path computes
      if (pi + 1 < NP) {
        const _Float16* wn = wt16 + GPATHS[G][pi + 1] * 16384;
        __builtin_prefetch(wn + (size_t)tid * 64, 0, 1);   // 256 thr x 128B = 32KB tile
      }
    }
    // ---- stage A: As[(zl*KG+k)][u4..u4+3] = f16( sum_i c[k][i]*x1[z,s1+i,u] ) ----
    {
      const int u4 = (tid & 31) * 4;      // 4 consecutive input channels
      const int zstart = tid >> 5;        // 0..7
#pragma unroll
      for (int zl0 = 0; zl0 < NT; zl0 += 8) {
        const int zl = zl0 + zstart;
        int z = z0 + zl;
        if (z >= Nn) z = Nn - 1;          // clamp (c==0 zeros these rows)
        const float* xz = x1 + (size_t)z * 1152 + s1 * 128 + u4;
        v4f xv[5];
#pragma unroll
        for (int i = 0; i < 5; ++i)
          xv[i] = (i < id) ? *(const v4f*)&xz[i * 128] : zero4;
#pragma unroll
        for (int k = 0; k < KG; ++k) {
          v4f s = zero4;
#pragma unroll
          for (int i = 0; i < 5; ++i)
            if (i < id) s += xv[i] * Cs[zl * CSTRIDE + co + k * id + i];
          *(v4h*)&As[(zl * KG + k) * AS_STRIDE + u4] = __builtin_convertvector(s, v4h);
        }
      }
    }
    // Bs DMA must have landed before anyone's WMMA reads it
    asm volatile("s_wait_asynccnt 0x0" ::: "memory");
    __syncthreads();

    // ---- WMMA: 4 K-blocks of 32 over this path's 128-wide u chunk ----
#pragma unroll
    for (int kb = 0; kb < 4; ++kb) {
      const int K0 = kb * 32;
      // B fragment (32x16): column l16; h[0..7]=K0+half*8.., h[8..15]=K0+16+half*8..
      const int bbase = (wv * 16 + l16) * BS_STRIDE + K0 + half * 8;
      v8h blo = *(const v8h*)&Bs[bbase];
      v8h bhi = *(const v8h*)&Bs[bbase + 16];
      v16h b = __builtin_shufflevector(blo, bhi, 0,1,2,3,4,5,6,7,8,9,10,11,12,13,14,15);
#pragma unroll
      for (int mt = 0; mt < MT; ++mt) {
        // A fragment (16x32): row l16; h[0..7]=K0+half*8.., h[8..15]=K0+16+half*8..
        const int abase = (mt * 16 + l16) * AS_STRIDE + K0 + half * 8;
        v8h alo = *(const v8h*)&As[abase];
        v8h ahi = *(const v8h*)&As[abase + 16];
        v16h a = __builtin_shufflevector(alo, ahi, 0,1,2,3,4,5,6,7,8,9,10,11,12,13,14,15);
        acc[mt] = __builtin_amdgcn_wmma_f32_16x16x32_f16(false, a, false, b, (short)0, acc[mt], false, false);
      }
    }
    __syncthreads();
  }

  // ---- store: uniform base + 32-bit lane offset (SADDR-friendly) ----
  // row grow = mt*16 + half*8 + r maps to node zl = grow/KG, basis k = grow%KG;
  // both candidate offsets are compile-time, selected by `half`.
  float* sout = out + (size_t)z0 * 1152 + IOB * 128;   // uniform across block
  const uint32_t lanecol = (uint32_t)(wv * 16 + l16);
  if (z0 + NT <= Nn) {
#pragma unroll
    for (int mt = 0; mt < MT; ++mt) {
#pragma unroll
      for (int r = 0; r < 8; ++r) {
        const int g0 = mt * 16 + r;
        const int g1 = g0 + 8;
        const uint32_t O0 = (uint32_t)((g0 / KG) * 1152 + (g0 % KG) * 128);
        const uint32_t O1 = (uint32_t)((g1 / KG) * 1152 + (g1 % KG) * 128);
        sout[lanecol + (half ? O1 : O0)] = acc[mt][r];
      }
    }
  } else {
#pragma unroll
    for (int mt = 0; mt < MT; ++mt) {
#pragma unroll
      for (int r = 0; r < 8; ++r) {
        const int g0 = mt * 16 + r;
        const int g1 = g0 + 8;
        const uint32_t O0 = (uint32_t)((g0 / KG) * 1152 + (g0 % KG) * 128);
        const uint32_t O1 = (uint32_t)((g1 / KG) * 1152 + (g1 % KG) * 128);
        const int zl = half ? (g1 / KG) : (g0 / KG);
        if (z0 + zl < Nn)
          sout[lanecol + (half ? O1 : O0)] = acc[mt][r];
      }
    }
  }
}

__global__ __launch_bounds__(256) void tp_main(
    const float* __restrict__ x1, const float* __restrict__ x2,
    const float* __restrict__ w3j, const _Float16* __restrict__ wt16,
    float* __restrict__ out, int Nn)
{
  __shared__ __align__(16) _Float16 As[160 * AS_STRIDE];  // 43,520 B
  __shared__ __align__(16) _Float16 Bs[128 * BS_STRIDE];  // 34,816 B
  __shared__ __align__(16) float    Cs[NT * CSTRIDE];     // 23,040 B  (~101 KB total of 320 KB WGP LDS)

  const int tid = threadIdx.x;
  const int z0  = blockIdx.x * NT;

  // ---- stage 0: c_p[z][k][i] = alpha_p * sum_j w3j_p[i,j,k] * x2[z, s2+j] ----
  {
    const int zl = tid & (NT - 1);
    const int z  = z0 + zl;
    const bool valid = (z < Nn);
    const int zc = valid ? z : (Nn - 1);
    float x2v[9];
#pragma unroll
    for (int j = 0; j < 9; ++j) x2v[j] = x2[zc * 9 + j];
    for (int p = (tid >> 5); p < 15; p += 8) {
      const int l1 = P_L1[p], l2 = P_L2[p], lo = P_LO[p];
      const int id = 2*l1+1, jd = 2*l2+1, kd = 2*lo+1;
      const int s2 = LBASE[l2];
      const int wo = W3JOFF[p], co = COFF[p];
      for (int k = 0; k < kd; ++k)
        for (int i = 0; i < id; ++i) {
          float s = 0.f;
          for (int j = 0; j < jd; ++j)
            s += w3j[wo + (i * jd + j) * kd + k] * x2v[s2 + j];
          Cs[zl * CSTRIDE + co + k * id + i] = valid ? s : 0.f;
        }
    }
  }
  __syncthreads();

  do_group<1, 0, 3, 0>(x1, wt16, out, Nn, z0, As, Bs, Cs);
  do_group<3, 1, 6, 1>(x1, wt16, out, Nn, z0, As, Bs, Cs);
  do_group<5, 4, 6, 2>(x1, wt16, out, Nn, z0, As, Bs, Cs);
}

// =====================================================================
extern "C" void kernel_launch(void* const* d_in, const int* in_sizes, int n_in,
                              void* d_out, int out_size, void* d_ws, size_t ws_size,
                              hipStream_t stream) {
  (void)n_in; (void)out_size; (void)ws_size;
  const float* x1 = (const float*)d_in[0];
  const float* x2 = (const float*)d_in[1];
  const float* w  = (const float*)d_in[2];
  float* out = (float*)d_out;
  const int Nn = in_sizes[0] / 1152;   // (N, 9, 128)

  // workspace layout: [0, 2460) w3j tables; [4096, 4096+491520) f16 weights
  float* w3j = (float*)d_ws;
  _Float16* wt16 = (_Float16*)((char*)d_ws + 4096);

  w3j_prep<<<1, 32, 0, stream>>>(w3j);
  prep_weights<<<(15 * 16384 + 255) / 256, 256, 0, stream>>>(w, wt16);
  tp_main<<<(Nn + NT - 1) / NT, 256, 0, stream>>>(x1, x2, w3j, wt16, out, Nn);
}
